// FragmentDecoder_40587440947458
// MI455X (gfx1250) — matrix operand order repaired and checked by reading, not compiled, not run on hardware
//
#include <hip/hip_runtime.h>

// FragmentDecoder for MI455X (gfx1250): all GEMMs via v_wmma_f32_16x16x32_bf16
// (bf16 A/B, fp32 accum). Each wave computes a 32x64 C tile (2 A frags x 4 B
// frags -> 8 WMMAs per 12 b128 loads). Softmax / LayerNorm / spatial-bias MLP
// in fp32 VALU. Workspace usage ~288 MB carved from d_ws.

#define B_     8
#define LR_    512
#define M_     512
#define HID_   512
#define NH_    8
#define HD_    64
#define NL_    4
#define VOCAB_ 4096
#define BINS_  72

typedef unsigned short u16;
typedef __attribute__((ext_vector_type(16))) __bf16 v16bf;
typedef __attribute__((ext_vector_type(8)))  float  v8f;

union V16U { uint4 q[2]; v16bf v; };

__device__ __forceinline__ u16 f2bf(float f) {
  unsigned u = __float_as_uint(f);
  unsigned r = u + 0x7FFFu + ((u >> 16) & 1u);
  if ((u & 0x7F800000u) == 0x7F800000u) r = u;  // Inf/NaN passthrough
  return (u16)(r >> 16);
}
__device__ __forceinline__ float gelu_f(float x) {
  return 0.5f * x * (1.0f + erff(x * 0.70710678118654752f));
}
__device__ __forceinline__ float n2n(float v) {
  if (isnan(v)) return 0.0f;
  if (isinf(v)) return v > 0.0f ? 3.4028235e38f : -3.4028235e38f;
  return v;
}

// ---------------------------------------------------------------------------
// WMMA GEMM: C[z] = act(alpha * A[z] (MxK) * B[z]^T (B stored NxK) + bias)
// Block = 4 waves along M; each wave -> 32 rows x 64 cols (2x4 sub-tiles).
// A layout: row=lane&15, half=lane>>4; elems 0..15 = K k0+16h+i,
// same k order as B.  B: col=lane&15, elems 0..15 = K k0+16h+i.
// C f32: VGPR r -> M = base + 8*half + r, N = base + (lane&15).
// Out-of-range N columns: B row pointer clamped to row 0 (garbage only lands
// in the clamped lanes' own D columns, which are never stored). No divergent
// branches before the WMMAs, so EXEC stays all-ones.
// ---------------------------------------------------------------------------
__global__ __launch_bounds__(128)
void gemm_bf16_wmma(const u16* __restrict__ A, const u16* __restrict__ Bm,
                    void* __restrict__ Cout, const float* __restrict__ bias,
                    int M, int N, int K, int lda, int ldb, int ldc,
                    long sA, long sB, long csOuter, long csInner, int zdiv,
                    float alpha, int doGelu, int outBf)
{
  int lane = threadIdx.x & 31;
  int wv   = threadIdx.x >> 5;
  int mt   = blockIdx.y * 4 + wv;          // 32-row tile index
  if (mt * 32 >= M) return;                // wave-uniform
  int z = blockIdx.z;

  const u16* Ab = A  + (long)z * sA + (long)(mt * 32) * lda;
  const u16* Bb = Bm + (long)z * sB;
  long cbase = (long)(z / zdiv) * csOuter + (long)(z % zdiv) * csInner;

  int row  = lane & 15;
  int half = lane >> 4;
  const u16* arow0 = Ab + (long)row * lda + 16 * half;     // rows mt*32 + 0..15
  const u16* arow1 = arow0 + (long)16 * lda;               // rows mt*32 + 16..31

  int n0 = blockIdx.x * 64 + row;          // sub-tile j covers column n0 + j*16
  int nc0 = n0, nc1 = n0 + 16, nc2 = n0 + 32, nc3 = n0 + 48;
  const u16* br0 = Bb + (long)(nc0 < N ? nc0 : 0) * ldb + 16 * half;
  const u16* br1 = Bb + (long)(nc1 < N ? nc1 : 0) * ldb + 16 * half;
  const u16* br2 = Bb + (long)(nc2 < N ? nc2 : 0) * ldb + 16 * half;
  const u16* br3 = Bb + (long)(nc3 < N ? nc3 : 0) * ldb + 16 * half;

  v8f p00, p01, p02, p03;   // rows 0..15  x sub-tiles 0..3
  v8f p10, p11, p12, p13;   // rows 16..31 x sub-tiles 0..3
#pragma unroll
  for (int i = 0; i < 8; ++i) {
    p00[i]=0.f; p01[i]=0.f; p02[i]=0.f; p03[i]=0.f;
    p10[i]=0.f; p11[i]=0.f; p12[i]=0.f; p13[i]=0.f;
  }

  for (int k0 = 0; k0 < K; k0 += 32) {
    V16U a0, a1, b0, b1, b2, b3;
    a0.q[0] = *(const uint4*)(arow0 + k0);
    a0.q[1] = *(const uint4*)(arow0 + k0 + 8);
    a1.q[0] = *(const uint4*)(arow1 + k0);
    a1.q[1] = *(const uint4*)(arow1 + k0 + 8);
    b0.q[0] = *(const uint4*)(br0 + k0);
    b0.q[1] = *(const uint4*)(br0 + k0 + 8);
    b1.q[0] = *(const uint4*)(br1 + k0);
    b1.q[1] = *(const uint4*)(br1 + k0 + 8);
    b2.q[0] = *(const uint4*)(br2 + k0);
    b2.q[1] = *(const uint4*)(br2 + k0 + 8);
    b3.q[0] = *(const uint4*)(br3 + k0);
    b3.q[1] = *(const uint4*)(br3 + k0 + 8);
    p00 = __builtin_amdgcn_wmma_f32_16x16x32_bf16(false, a0.v, false, b0.v,
                                                  (short)0, p00, false, false);
    p10 = __builtin_amdgcn_wmma_f32_16x16x32_bf16(false, a1.v, false, b0.v,
                                                  (short)0, p10, false, false);
    p01 = __builtin_amdgcn_wmma_f32_16x16x32_bf16(false, a0.v, false, b1.v,
                                                  (short)0, p01, false, false);
    p11 = __builtin_amdgcn_wmma_f32_16x16x32_bf16(false, a1.v, false, b1.v,
                                                  (short)0, p11, false, false);
    p02 = __builtin_amdgcn_wmma_f32_16x16x32_bf16(false, a0.v, false, b2.v,
                                                  (short)0, p02, false, false);
    p12 = __builtin_amdgcn_wmma_f32_16x16x32_bf16(false, a1.v, false, b2.v,
                                                  (short)0, p12, false, false);
    p03 = __builtin_amdgcn_wmma_f32_16x16x32_bf16(false, a0.v, false, b3.v,
                                                  (short)0, p03, false, false);
    p13 = __builtin_amdgcn_wmma_f32_16x16x32_bf16(false, a1.v, false, b3.v,
                                                  (short)0, p13, false, false);
  }

  // epilogue: bias + optional GELU, store valid columns
#pragma unroll
  for (int j = 0; j < 4; ++j) {
    int n = n0 + j * 16;
    if (n >= N) continue;
    const v8f& accLo = (j == 0) ? p00 : (j == 1) ? p01 : (j == 2) ? p02 : p03;
    const v8f& accHi = (j == 0) ? p10 : (j == 1) ? p11 : (j == 2) ? p12 : p13;
    float bv = bias ? bias[n] : 0.0f;
#pragma unroll
    for (int r = 0; r < 8; ++r) {
      int m = mt * 32 + 8 * half + r;
      float v = accLo[r] * alpha + bv;
      if (doGelu) v = gelu_f(v);
      long idx = cbase + (long)m * ldc + n;
      if (outBf) ((u16*)Cout)[idx] = f2bf(v);
      else       ((float*)Cout)[idx] = v;
    }
#pragma unroll
    for (int r = 0; r < 8; ++r) {
      int m = mt * 32 + 16 + 8 * half + r;
      float v = accHi[r] * alpha + bv;
      if (doGelu) v = gelu_f(v);
      long idx = cbase + (long)m * ldc + n;
      if (outBf) ((u16*)Cout)[idx] = f2bf(v);
      else       ((float*)Cout)[idx] = v;
    }
  }
}

// ---------------------------------------------------------------------------
__global__ void cvt_f32_bf16_kernel(const float* __restrict__ src,
                                    u16* __restrict__ dst, long n) {
  long i = (long)blockIdx.x * blockDim.x + threadIdx.x;
  if (i < n) dst[i] = f2bf(src[i]);
}

__global__ void transpose_cvt_kernel(const float* __restrict__ src,
                                     u16* __restrict__ dst, int K, int N) {
  long i = (long)blockIdx.x * blockDim.x + threadIdx.x;
  if (i >= (long)K * N) return;
  int k = (int)(i / N), n = (int)(i % N);
  dst[(long)n * K + k] = f2bf(src[i]);
}

// x[b,m,:] = emb_table[tok] + posenc(m,:)
__global__ void embed_kernel(const float* __restrict__ emb,
                             const int* __restrict__ tf,
                             float* __restrict__ xf, u16* __restrict__ xb) {
  long i = (long)blockIdx.x * blockDim.x + threadIdx.x;
  if (i >= (long)B_ * M_ * HID_) return;
  int d = (int)(i % HID_);
  long bm = i / HID_;
  int m = (int)(bm % M_);
  int tok = tf[bm];
  int j = d >> 1;
  float freq = expf(-(float)(2 * j) * (9.210340371976184f / (float)HID_)); // ln(1e4)
  float ang = (float)m * freq;
  float pe = (d & 1) ? cosf(ang) : sinf(ang);
  float v = emb[(long)tok * HID_ + d] + pe;
  xf[i] = v;
  xb[i] = f2bf(v);
}

// per (b,m): clamp frag idx, nan_to_num coords, compute CA position and v_dir
__global__ void prep_kernel(const float* __restrict__ bb,
                            const int* __restrict__ fri,
                            float* __restrict__ cav, float* __restrict__ vdir) {
  int i = blockIdx.x * blockDim.x + threadIdx.x;
  if (i >= B_ * M_) return;
  int b = i / M_;
  int idx = fri[i];
  idx = idx < 0 ? 0 : (idx > LR_ - 1 ? LR_ - 1 : idx);
  const float* p = bb + ((long)b * LR_ + idx) * 9;
  float c[9];
#pragma unroll
  for (int k = 0; k < 9; ++k) c[k] = n2n(p[k]);
  float a0 = c[0]-c[3], a1 = c[1]-c[4], a2 = c[2]-c[5];
  float na = fmaxf(sqrtf(a0*a0+a1*a1+a2*a2), 1e-12f);
  a0/=na; a1/=na; a2/=na;
  float b0 = c[6]-c[3], b1 = c[7]-c[4], b2 = c[8]-c[5];
  float nb = fmaxf(sqrtf(b0*b0+b1*b1+b2*b2), 1e-12f);
  b0/=nb; b1/=nb; b2/=nb;
  float s0 = a0+b0+1e-6f, s1 = a1+b1+1e-6f, s2 = a2+b2+1e-6f;
  float ns = fmaxf(sqrtf(s0*s0+s1*s1+s2*s2), 1e-12f);
  cav[(long)i*3+0] = c[3]; cav[(long)i*3+1] = c[4]; cav[(long)i*3+2] = c[5];
  vdir[(long)i*3+0] = s0/ns; vdir[(long)i*3+1] = s1/ns; vdir[(long)i*3+2] = s2/ns;
}

// per (b,i,j): RBF/cos/mask features -> 19->32 (gelu) ->8 MLP; + causal
__global__ __launch_bounds__(256)
void sbias_kernel(const float* __restrict__ cav, const float* __restrict__ vd,
                  const int* __restrict__ maskI,
                  const float* __restrict__ w1, const float* __restrict__ b1,
                  const float* __restrict__ w2, const float* __restrict__ b2,
                  float* __restrict__ outMask) {
  __shared__ float w1s[19*32], b1s[32], w2s[32*8], b2s[8];
  __shared__ float ci[3], vi[3], mi;
  int t = threadIdx.x;
  for (int idx = t; idx < 19*32; idx += 256) w1s[idx] = w1[idx];
  if (t < 32) b1s[t] = b1[t];
  if (t < 32*8) w2s[t] = w2[t];
  if (t < 8) b2s[t] = b2[t];
  int i = blockIdx.y, b = blockIdx.z;
  if (t == 0) {
    long pi = ((long)b * M_ + i) * 3;
    ci[0]=cav[pi]; ci[1]=cav[pi+1]; ci[2]=cav[pi+2];
    vi[0]=vd[pi];  vi[1]=vd[pi+1];  vi[2]=vd[pi+2];
    mi = maskI[b*M_+i] ? 1.0f : 0.0f;
  }
  __syncthreads();
  int j = blockIdx.x * 256 + t;
  if (j >= M_) return;
  long pj = ((long)b * M_ + j) * 3;
  float cj0=cav[pj], cj1=cav[pj+1], cj2=cav[pj+2];
  float vj0=vd[pj],  vj1=vd[pj+1],  vj2=vd[pj+2];
  float mj = maskI[b*M_+j] ? 1.0f : 0.0f;
  float dx=ci[0]-cj0, dy=ci[1]-cj1, dz=ci[2]-cj2;
  float D = sqrtf(dx*dx+dy*dy+dz*dz);
  float cosv = vi[0]*vj0 + vi[1]*vj1 + vi[2]*vj2;
  float feat[19];
#pragma unroll
  for (int k = 0; k < 16; ++k) {
    float mu = (float)k * (20.0f/15.0f);        // linspace(0,20,16)
    float e = (D - mu) * 0.8f;                  // / (20/16)
    feat[k] = expf(-e*e);
  }
  feat[16]=cosv; feat[17]=mi; feat[18]=mj;
  float h1[32];
#pragma unroll
  for (int o = 0; o < 32; ++o) {
    float s = b1s[o];
#pragma unroll
    for (int f = 0; f < 19; ++f) s += feat[f]*w1s[f*32+o];
    h1[o] = gelu_f(s);
  }
  float causal = (j > i) ? -1.0e9f : 0.0f;
#pragma unroll
  for (int h = 0; h < 8; ++h) {
    float s = b2s[h];
#pragma unroll
    for (int o = 0; o < 32; ++o) s += h1[o]*w2s[o*8+h];
    outMask[(((long)(b*8+h) * M_) + i) * M_ + j] = s + causal;
  }
}

// out = LN(x [+ res]) * g + b ; optional f32 and bf16 outputs
__global__ __launch_bounds__(128)
void ln_kernel(const float* __restrict__ x, const float* __restrict__ res,
               const float* __restrict__ g, const float* __restrict__ bt,
               float* __restrict__ outF, u16* __restrict__ outB) {
  __shared__ float red[128];
  int row = blockIdx.x, t = threadIdx.x;
  const float* xr = x + (long)row * HID_;
  const float* rr = res ? res + (long)row * HID_ : nullptr;
  float v[4];
  float s = 0.0f;
#pragma unroll
  for (int i = 0; i < 4; ++i) {
    int c = t + i * 128;
    float a = xr[c] + (rr ? rr[c] : 0.0f);
    v[i] = a; s += a;
  }
  red[t] = s; __syncthreads();
  for (int st = 64; st > 0; st >>= 1) { if (t < st) red[t] += red[t+st]; __syncthreads(); }
  float mean = red[0] * (1.0f / HID_);
  __syncthreads();
  float sq = 0.0f;
#pragma unroll
  for (int i = 0; i < 4; ++i) { float d0 = v[i] - mean; sq += d0*d0; }
  red[t] = sq; __syncthreads();
  for (int st = 64; st > 0; st >>= 1) { if (t < st) red[t] += red[t+st]; __syncthreads(); }
  float var = red[0] * (1.0f / HID_);
  float rstd = rsqrtf(var + 1e-5f);
#pragma unroll
  for (int i = 0; i < 4; ++i) {
    int c = t + i * 128;
    float o = (v[i] - mean) * rstd * g[c] + bt[c];
    if (outF) outF[(long)row * HID_ + c] = o;
    if (outB) outB[(long)row * HID_ + c] = f2bf(o);
  }
}

// softmax over K=512 with optional additive mask; bf16 output
__global__ __launch_bounds__(256)
void softmax_kernel(const float* __restrict__ s, const float* __restrict__ mask,
                    u16* __restrict__ out) {
  __shared__ float red[256];
  int q = blockIdx.x;
  long bh = blockIdx.y;
  long base = (bh * M_ + q) * M_;
  int t = threadIdx.x;
  float a0 = s[base + t], a1 = s[base + t + 256];
  if (mask) { a0 += mask[base + t]; a1 += mask[base + t + 256]; }
  red[t] = fmaxf(a0, a1); __syncthreads();
  for (int st = 128; st > 0; st >>= 1) { if (t < st) red[t] = fmaxf(red[t], red[t+st]); __syncthreads(); }
  float mx = red[0]; __syncthreads();
  float e0 = expf(a0 - mx), e1 = expf(a1 - mx);
  red[t] = e0 + e1; __syncthreads();
  for (int st = 128; st > 0; st >>= 1) { if (t < st) red[t] += red[t+st]; __syncthreads(); }
  float inv = 1.0f / red[0];
  out[base + t]       = f2bf(e0 * inv);
  out[base + t + 256] = f2bf(e1 * inv);
}

// split projections into per-head q/k [bh,m,64] and v transposed [bh,64,m]
__global__ void pack_kernel(const u16* __restrict__ qsrc, int qld,
                            const u16* __restrict__ ksrc, int kld, int koff,
                            const u16* __restrict__ vsrc, int vld, int voff,
                            u16* __restrict__ qb, u16* __restrict__ kb,
                            u16* __restrict__ vtb) {
  long i = (long)blockIdx.x * blockDim.x + threadIdx.x;
  if (i >= (long)B_ * M_ * HID_) return;
  int hd = (int)(i % HID_);
  long bm = i / HID_;
  int m = (int)(bm % M_);
  int b = (int)(bm / M_);
  int h = hd >> 6, d = hd & 63;
  long bh = (long)b * NH_ + h;
  qb[(bh * M_ + m) * HD_ + d] = qsrc[bm * qld + hd];
  kb[(bh * M_ + m) * HD_ + d] = ksrc[bm * kld + koff + hd];
  vtb[(bh * HD_ + d) * M_ + m] = vsrc[bm * vld + voff + hd];
}

// ---------------------------------------------------------------------------
extern "C" void kernel_launch(void* const* d_in, const int* in_sizes, int n_in,
                              void* d_out, int out_size, void* d_ws, size_t ws_size,
                              hipStream_t stream) {
  (void)in_sizes; (void)n_in; (void)out_size; (void)ws_size;
  const float* node      = (const float*)d_in[0];
  const float* bbc       = (const float*)d_in[1];
  const float* embt      = (const float*)d_in[2];
  const float* enc_w     = (const float*)d_in[3];
  const float* enc_b     = (const float*)d_in[4];
  const float* sa_in_w   = (const float*)d_in[5];
  const float* sa_in_b   = (const float*)d_in[6];
  const float* sa_out_w  = (const float*)d_in[7];
  const float* sa_out_b  = (const float*)d_in[8];
  const float* ca_in_w   = (const float*)d_in[9];
  const float* ca_in_b   = (const float*)d_in[10];
  const float* ca_out_w  = (const float*)d_in[11];
  const float* ca_out_b  = (const float*)d_in[12];
  const float* ff1_w     = (const float*)d_in[13];
  const float* ff1_b     = (const float*)d_in[14];
  const float* ff2_w     = (const float*)d_in[15];
  const float* ff2_b     = (const float*)d_in[16];
  const float* ln_g      = (const float*)d_in[17];
  const float* ln_b      = (const float*)d_in[18];
  const float* sb_w1     = (const float*)d_in[19];
  const float* sb_b1     = (const float*)d_in[20];
  const float* sb_w2     = (const float*)d_in[21];
  const float* sb_b2     = (const float*)d_in[22];
  const float* head_ln_g = (const float*)d_in[23];
  const float* head_ln_b = (const float*)d_in[24];
  const float* head_w1   = (const float*)d_in[25];
  const float* head_b1   = (const float*)d_in[26];
  const float* type_w2   = (const float*)d_in[27];
  const float* type_b2   = (const float*)d_in[28];
  const float* tb_w2     = (const float*)d_in[29];
  const float* tb_b2     = (const float*)d_in[30];
  const int*   tf        = (const int*)d_in[31];
  const int*   fri       = (const int*)d_in[32];
  const int*   fmask     = (const int*)d_in[33];

  char* ws = (char*)d_ws;
  size_t off = 0;
  auto alloc = [&](size_t bytes) -> void* {
    void* p = ws + off;
    off += (bytes + 255) & ~(size_t)255;
    return p;
  };

  const long BM = (long)B_ * M_;            // 4096
  const long BH = (long)B_ * NH_;           // 64

  // fp32 buffers
  float* x_f    = (float*)alloc(BM * HID_ * 4);
  float* scores = (float*)alloc(BH * M_ * M_ * 4);
  float* maskb  = (float*)alloc(BH * M_ * M_ * 4);
  float* y_f    = (float*)alloc(BM * HID_ * 4);
  float* cav    = (float*)alloc(BM * 3 * 4);
  float* vdr    = (float*)alloc(BM * 3 * 4);
  // bf16 buffers
  u16* node_bf = (u16*)alloc(BM * HID_ * 2);
  u16* mem_bf  = (u16*)alloc(BM * HID_ * 2);
  u16* x_bf    = (u16*)alloc(BM * HID_ * 2);
  u16* qkv_bf  = (u16*)alloc(BM * 1536 * 2);
  u16* q_bf    = (u16*)alloc(BH * M_ * HD_ * 2);
  u16* k_bf    = (u16*)alloc(BH * M_ * HD_ * 2);
  u16* vT_bf   = (u16*)alloc(BH * HD_ * M_ * 2);
  u16* a_bf    = (u16*)alloc(BH * M_ * M_ * 2);
  u16* attn_bf = (u16*)alloc(BM * HID_ * 2);
  u16* ffh_bf  = (u16*)alloc(BM * 2048 * 2);
  u16* hh_bf   = (u16*)alloc(BM * HID_ * 2);
  // bf16 weights
  u16* w_enc    = (u16*)alloc(512L*512*2);
  u16* w_sa_in  = (u16*)alloc(4L*1536*512*2);
  u16* w_sa_out = (u16*)alloc(4L*512*512*2);
  u16* w_ca_in  = (u16*)alloc(4L*1536*512*2);
  u16* w_ca_out = (u16*)alloc(4L*512*512*2);
  u16* w_ff1    = (u16*)alloc(4L*2048*512*2);
  u16* w_ff2    = (u16*)alloc(4L*512*2048*2);
  u16* w_h1     = (u16*)alloc(3L*512*512*2);
  u16* w_type   = (u16*)alloc(4096L*512*2);
  u16* w_tb     = (u16*)alloc(2L*72*512*2);

  auto CVT = [&](const float* s, u16* d, long n) {
    cvt_f32_bf16_kernel<<<(unsigned)((n + 255) / 256), 256, 0, stream>>>(s, d, n);
  };
  auto TR = [&](const float* s, u16* d, int K, int N) {
    long n = (long)K * N;
    transpose_cvt_kernel<<<(unsigned)((n + 255) / 256), 256, 0, stream>>>(s, d, K, N);
  };
  auto GEMM = [&](const u16* A, const u16* Bmat, void* C, const float* bias,
                  int M, int N, int K, int lda, int ldb, int ldc,
                  long sA, long sB, long cso, long csi, int zdiv, int batch,
                  float alpha, int doGelu, int outBf) {
    dim3 grid((N + 63) / 64, (M + 127) / 128, batch);
    gemm_bf16_wmma<<<grid, 128, 0, stream>>>(A, Bmat, C, bias, M, N, K,
        lda, ldb, ldc, sA, sB, cso, csi, zdiv, alpha, doGelu, outBf);
  };

  // ---- weight conversion / transposition (bf16) ----
  CVT(node, node_bf, BM * HID_);
  TR(enc_w, w_enc, 512, 512);
  CVT(sa_in_w,  w_sa_in,  4L*1536*512);
  CVT(sa_out_w, w_sa_out, 4L*512*512);
  CVT(ca_in_w,  w_ca_in,  4L*1536*512);
  CVT(ca_out_w, w_ca_out, 4L*512*512);
  for (int l = 0; l < NL_; ++l) {
    TR(ff1_w + (long)l*512*2048, w_ff1 + (long)l*2048*512, 512, 2048);
    TR(ff2_w + (long)l*2048*512, w_ff2 + (long)l*512*2048, 2048, 512);
  }
  for (int i = 0; i < 3; ++i)
    TR(head_w1 + (long)i*512*512, w_h1 + (long)i*512*512, 512, 512);
  TR(type_w2, w_type, 512, 4096);
  for (int i = 0; i < 2; ++i)
    TR(tb_w2 + (long)i*512*72, w_tb + (long)i*72*512, 512, 72);

  // ---- encoder: mem = node @ enc_w + enc_b  (bf16 out) ----
  GEMM(node_bf, w_enc, mem_bf, enc_b, (int)BM, 512, 512, 512, 512, 512,
       0, 0, 0, 0, 1, 1, 1.0f, 0, 1);

  // ---- token embedding + positional encoding ----
  embed_kernel<<<(unsigned)((BM * HID_ + 255) / 256), 256, 0, stream>>>(
      embt, tf, x_f, x_bf);

  // ---- spatial bias + causal mask ----
  prep_kernel<<<(unsigned)((BM + 255) / 256), 256, 0, stream>>>(bbc, fri, cav, vdr);
  sbias_kernel<<<dim3(M_/256, M_, B_), 256, 0, stream>>>(
      cav, vdr, fmask, sb_w1, sb_b1, sb_w2, sb_b2, maskb);

  const long sQK  = (long)M_ * HD_;       // 32768
  const long sSc  = (long)M_ * M_;        // 262144

  for (int l = 0; l < NL_; ++l) {
    // ======== self attention ========
    GEMM(x_bf, w_sa_in + (long)l*1536*512, qkv_bf, sa_in_b + (long)l*1536,
         (int)BM, 1536, 512, 512, 512, 1536, 0, 0, 0, 0, 1, 1, 1.0f, 0, 1);
    pack_kernel<<<(unsigned)((BM * HID_ + 255) / 256), 256, 0, stream>>>(
        qkv_bf, 1536, qkv_bf, 1536, 512, qkv_bf, 1536, 1024, q_bf, k_bf, vT_bf);
    // scores = q k^T / 8
    GEMM(q_bf, k_bf, scores, nullptr, M_, M_, HD_, HD_, HD_, M_,
         sQK, sQK, sSc, 0, 1, (int)BH, 0.125f, 0, 0);
    softmax_kernel<<<dim3(M_, (unsigned)BH), 256, 0, stream>>>(scores, maskb, a_bf);
    // o = a v  ->  attn_bf laid out [b,m,h*64+d]
    GEMM(a_bf, vT_bf, attn_bf, nullptr, M_, HD_, M_, M_, M_, 512,
         sSc, sQK, (long)M_*HID_, HD_, NH_, (int)BH, 1.0f, 0, 1);
    GEMM(attn_bf, w_sa_out + (long)l*512*512, y_f, sa_out_b + (long)l*512,
         (int)BM, 512, 512, 512, 512, 512, 0, 0, 0, 0, 1, 1, 1.0f, 0, 0);
    ln_kernel<<<(unsigned)BM, 128, 0, stream>>>(
        x_f, y_f, ln_g + ((long)l*3 + 0)*512, ln_b + ((long)l*3 + 0)*512, x_f, x_bf);

    // ======== cross attention ========
    const u16* caw = w_ca_in + (long)l*1536*512;
    u16* qb32 = qkv_bf;                 // [BM,512]
    u16* kv32 = qkv_bf + BM * 512;      // [BM,1024]
    GEMM(x_bf, caw, qb32, ca_in_b + (long)l*1536,
         (int)BM, 512, 512, 512, 512, 512, 0, 0, 0, 0, 1, 1, 1.0f, 0, 1);
    GEMM(mem_bf, caw + 512L*512, kv32, ca_in_b + (long)l*1536 + 512,
         (int)BM, 1024, 512, 512, 512, 1024, 0, 0, 0, 0, 1, 1, 1.0f, 0, 1);
    pack_kernel<<<(unsigned)((BM * HID_ + 255) / 256), 256, 0, stream>>>(
        qb32, 512, kv32, 1024, 0, kv32, 1024, 512, q_bf, k_bf, vT_bf);
    GEMM(q_bf, k_bf, scores, nullptr, M_, M_, HD_, HD_, HD_, M_,
         sQK, sQK, sSc, 0, 1, (int)BH, 0.125f, 0, 0);
    softmax_kernel<<<dim3(M_, (unsigned)BH), 256, 0, stream>>>(scores, nullptr, a_bf);
    GEMM(a_bf, vT_bf, attn_bf, nullptr, M_, HD_, M_, M_, M_, 512,
         sSc, sQK, (long)M_*HID_, HD_, NH_, (int)BH, 1.0f, 0, 1);
    GEMM(attn_bf, w_ca_out + (long)l*512*512, y_f, ca_out_b + (long)l*512,
         (int)BM, 512, 512, 512, 512, 512, 0, 0, 0, 0, 1, 1, 1.0f, 0, 0);
    ln_kernel<<<(unsigned)BM, 128, 0, stream>>>(
        x_f, y_f, ln_g + ((long)l*3 + 1)*512, ln_b + ((long)l*3 + 1)*512, x_f, x_bf);

    // ======== FFN ========
    GEMM(x_bf, w_ff1 + (long)l*2048*512, ffh_bf, ff1_b + (long)l*2048,
         (int)BM, 2048, 512, 512, 512, 2048, 0, 0, 0, 0, 1, 1, 1.0f, 1, 1);
    GEMM(ffh_bf, w_ff2 + (long)l*512*2048, y_f, ff2_b + (long)l*512,
         (int)BM, 512, 2048, 2048, 2048, 512, 0, 0, 0, 0, 1, 1, 1.0f, 0, 0);
    ln_kernel<<<(unsigned)BM, 128, 0, stream>>>(
        x_f, y_f, ln_g + ((long)l*3 + 2)*512, ln_b + ((long)l*3 + 2)*512, x_f, x_bf);
  }

  // ======== output heads ========
  float* outp = (float*)d_out;
  for (int i = 0; i < 3; ++i) {
    ln_kernel<<<(unsigned)BM, 128, 0, stream>>>(
        x_f, nullptr, head_ln_g + (long)i*512, head_ln_b + (long)i*512,
        nullptr, hh_bf);
    GEMM(hh_bf, w_h1 + (long)i*512*512, attn_bf, head_b1 + (long)i*512,
         (int)BM, 512, 512, 512, 512, 512, 0, 0, 0, 0, 1, 1, 1.0f, 1, 1);
    if (i == 0) {
      GEMM(attn_bf, w_type, outp, type_b2,
           (int)BM, VOCAB_, 512, 512, 512, VOCAB_, 0, 0, 0, 0, 1, 1, 1.0f, 0, 0);
    } else {
      float* dst = outp + (long)BM * VOCAB_ + (long)(i - 1) * BM * BINS_;
      GEMM(attn_bf, w_tb + (long)(i - 1)*72*512, dst, tb_b2 + (long)(i - 1)*72,
           (int)BM, BINS_, 512, 512, 512, BINS_, 0, 0, 0, 0, 1, 1, 1.0f, 0, 0);
    }
  }
}
